// TritonVisionProjector_26998164423217
// MI455X (gfx1250) — matrix operand order repaired and checked by reading, not compiled, not run on hardware
//
#include <hip/hip_runtime.h>
#include <math.h>

// ---------------------------------------------------------------------------
// W8A16 two-layer MLP for MI455X (gfx1250, wave32, WMMA).
//   L1: h   = gelu(x @ W1^T + b1)   x:f32 [N,IN],  W1:int8 [HID,IN]  -> h bf16
//   L2: out = h @ W2^T + b2         h:bf16 [N,HID], W2:int8 [OUT,HID] -> f32
// Compute-bound (773 GFLOP vs ~700MB HBM) => bf16 WMMA, f32 accumulate.
// Double-buffered LDS (1 barrier/iter); layer-2 A panel copied with
// GLOBAL_LOAD_ASYNC_TO_LDS_B128 (ASYNCcnt) so the VALU stays free for WMMA.
// ---------------------------------------------------------------------------

typedef __bf16 v8bf  __attribute__((ext_vector_type(8)));
typedef __bf16 v16bf __attribute__((ext_vector_type(16)));
typedef float  v8f   __attribute__((ext_vector_type(8)));
typedef float  vf4   __attribute__((ext_vector_type(4)));

static constexpr int NROWS = 18432;
static constexpr int IN1   = 1024;
static constexpr int HID1  = 4096;
static constexpr int OUT1  = 4096;

static constexpr int BM = 256;       // block rows (8 waves x 32)
static constexpr int BN = 64;        // block cols (4 x 16-wide WMMA tiles)
static constexpr int BK = 32;        // K step (one bf16 WMMA K)
static constexpr int LDSK = BK + 8;  // 40 halves = 80 B row stride (16B aligned)

union AFragU { v16bf v; v8bf h[2]; };

// raw LDS byte offset of a __shared__ object (flat addr low 32 bits, ISA §10.2)
__device__ __forceinline__ unsigned lds_off(const void* p) {
    return (unsigned)(unsigned long long)p;
}

template <bool A_IS_F32, bool DO_GELU, bool OUT_BF16>
__global__ __launch_bounds__(256, 2)
void w8a16_gemm(const void* __restrict__ Aglob,        // [M,K] f32 or bf16
                const signed char* __restrict__ Wq,    // [Ncols,K] int8 row-major
                const float* __restrict__ wscale,      // [Ncols]
                const float* __restrict__ bias,        // [Ncols]
                void* __restrict__ Out,                // [M,Ncols] bf16 or f32
                int K, int Ncols)
{
    __shared__ __attribute__((aligned(16))) __bf16 As[2][BM][LDSK];
    __shared__ __attribute__((aligned(16))) __bf16 Bs[2][BN][LDSK];

    const int tid  = threadIdx.x;
    const int lane = tid & 31;
    const int wave = tid >> 5;          // 0..7
    const int hf   = lane >> 4;         // 0: lanes 0-15, 1: lanes 16-31
    const int lrow = lane & 15;

    const int rowBase = blockIdx.y * BM;
    const int colBase = blockIdx.x * BN;

    // staging coordinates: A -> one full 32-wide K row per thread,
    // W -> 8 int8 per thread (64 cols x 32 K tile).
    const int arow = tid;
    const int wcol = tid >> 2;
    const int wkb  = (tid & 3) * 8;
    const float wsc = wscale[colBase + wcol];

    const float*  Af32 = (const float*)Aglob;
    const __bf16* Abf  = (const __bf16*)Aglob;

    // ---- staging helpers -------------------------------------------------
    auto stage_a_f32 = [&](int buf, const vf4* pf) {
        #pragma unroll
        for (int q = 0; q < 4; ++q) {
            v8bf hv;
            #pragma unroll
            for (int e = 0; e < 4; ++e) {
                hv[e]     = (__bf16)pf[2 * q][e];
                hv[4 + e] = (__bf16)pf[2 * q + 1][e];
            }
            *(v8bf*)&As[buf][arow][q * 8] = hv;
        }
    };
    auto issue_a_async = [&](int buf, int k0) {
        const __bf16* ap = Abf + (size_t)(rowBase + arow) * K + k0;
        #pragma unroll
        for (int q = 0; q < 4; ++q) {
            unsigned           dst = lds_off(&As[buf][arow][q * 8]);
            unsigned long long src = (unsigned long long)(ap + q * 8);
            asm volatile("global_load_async_to_lds_b128 %0, %1, off"
                         :: "v"(dst), "v"(src) : "memory");
        }
    };
    auto stage_w = [&](int buf, uint2 raw) {
        v8bf wb;
        #pragma unroll
        for (int e = 0; e < 8; ++e) {
            unsigned int word = (e < 4) ? raw.x : raw.y;
            signed char  b    = (signed char)((word >> ((e & 3) * 8)) & 0xffu);
            wb[e] = (__bf16)((float)b * wsc);
        }
        *(v8bf*)&Bs[buf][wcol][wkb] = wb;
    };

    // ---- accumulators: 2 (M) x 4 (N) tiles of 16x16 f32 ------------------
    v8f acc[2][4];
    #pragma unroll
    for (int mi = 0; mi < 2; ++mi)
        #pragma unroll
        for (int j = 0; j < 4; ++j)
            #pragma unroll
            for (int e = 0; e < 8; ++e)
                acc[mi][j][e] = 0.0f;

    const int niter = K / BK;

    // ---- prologue: fill buffer 0 -----------------------------------------
    if constexpr (A_IS_F32) {
        const float* ap = Af32 + (size_t)(rowBase + arow) * K;
        vf4 pf[8];
        #pragma unroll
        for (int q = 0; q < 8; ++q) pf[q] = *(const vf4*)(ap + q * 4);
        stage_a_f32(0, pf);
    } else {
        issue_a_async(0, 0);
    }
    stage_w(0, *(const uint2*)(Wq + (size_t)(colBase + wcol) * K + wkb));

    // ---- main loop: one barrier per K step, ping-pong buffers ------------
    for (int i = 0; i < niter; ++i) {
        const int cur = i & 1;
        const int nxt = cur ^ 1;
        const bool hasNext = (i + 1) < niter;

        if constexpr (!A_IS_F32)
            asm volatile("s_wait_asynccnt 0x0" ::: "memory");
        __syncthreads();   // buffer `cur` (incl. async writes) visible to all

        // issue next-tile global traffic before touching the WMMA pipe
        vf4  pfa[8];
        uint2 pfw;
        if (hasNext) {
            const int k0 = (i + 1) * BK;
            if constexpr (A_IS_F32) {
                const float* ap = Af32 + (size_t)(rowBase + arow) * K + k0;
                #pragma unroll
                for (int q = 0; q < 8; ++q) pfa[q] = *(const vf4*)(ap + q * 4);
            } else {
                issue_a_async(nxt, k0);      // DMA straight into idle buffer
            }
            pfw = *(const uint2*)(Wq + (size_t)(colBase + wcol) * K + k0 + wkb);
        }

        // fragments from buffer `cur`
        // A 16x32 bf16 per-lane layout: lanes 0-15 K{0..7,16..23} of row M=lane;
        // lanes 16-31 K{8..15,24..31} of row M=lane-16.
        AFragU afr[2];
        #pragma unroll
        for (int mi = 0; mi < 2; ++mi) {
            const int r = wave * 32 + mi * 16 + lrow;
            afr[mi].h[0] = *(const v8bf*)&As[cur][r][hf * 8];
            afr[mi].h[1] = *(const v8bf*)&As[cur][r][16 + hf * 8];
        }
        #pragma unroll
        for (int j = 0; j < 4; ++j) {
            // B 32x16 bf16: lanes 0-15 K{0..7,16..23}, lanes 16-31 K{8..15,24..31}, N = lane%16.
            AFragU bfr;
            const int c = j * 16 + lrow;
            bfr.h[0] = *(const v8bf*)&Bs[cur][c][hf * 8];
            bfr.h[1] = *(const v8bf*)&Bs[cur][c][16 + hf * 8];
            #pragma unroll
            for (int mi = 0; mi < 2; ++mi) {
                acc[mi][j] = __builtin_amdgcn_wmma_f32_16x16x32_bf16(
                    false, afr[mi].v, false, bfr.v,
                    (short)0, acc[mi][j], false, false);
            }
        }

        // stage next tile into the idle buffer (other waves may still read cur)
        if (hasNext) {
            if constexpr (A_IS_F32) stage_a_f32(nxt, pfa);
            stage_w(nxt, pfw);
        }
    }

    // ---- epilogue: bias (+gelu), store -----------------------------------
    // C/D layout: VGPR r -> M = r (lanes 0-15) / 8+r (lanes 16-31), N = lane%16.
    #pragma unroll
    for (int j = 0; j < 4; ++j) {
        const int colg = colBase + j * 16 + lrow;
        const float bv = bias[colg];
        #pragma unroll
        for (int mi = 0; mi < 2; ++mi) {
            #pragma unroll
            for (int r = 0; r < 8; ++r) {
                const int rowg = rowBase + wave * 32 + mi * 16 + hf * 8 + r;
                float v = acc[mi][j][r] + bv;
                if (DO_GELU)
                    v = 0.5f * v * (1.0f + erff(v * 0.70710678118654752f));
                if (OUT_BF16)
                    ((__bf16*)Out)[(size_t)rowg * Ncols + colg] = (__bf16)v;
                else
                    ((float*)Out)[(size_t)rowg * Ncols + colg] = v;
            }
        }
    }
}

extern "C" void kernel_launch(void* const* d_in, const int* in_sizes, int n_in,
                              void* d_out, int out_size, void* d_ws, size_t ws_size,
                              hipStream_t stream)
{
    const float*       x   = (const float*)d_in[0];
    const signed char* w1q = (const signed char*)d_in[1];
    const float*       w1s = (const float*)d_in[2];
    const float*       b1  = (const float*)d_in[3];
    const signed char* w2q = (const signed char*)d_in[4];
    const float*       w2s = (const float*)d_in[5];
    const float*       b2  = (const float*)d_in[6];

    __bf16* h = (__bf16*)d_ws;             // [NROWS, HID1] bf16 intermediate

    dim3 blk(256, 1, 1);
    dim3 g1(HID1 / BN, NROWS / BM, 1);     // 64 x 72
    w8a16_gemm<true, true, true><<<g1, blk, 0, stream>>>(
        x, w1q, w1s, b1, h, IN1, HID1);

    dim3 g2(OUT1 / BN, NROWS / BM, 1);     // 64 x 72
    w8a16_gemm<false, false, false><<<g2, blk, 0, stream>>>(
        h, w2q, w2s, b2, d_out, HID1, OUT1);
}
